// GTN_60318520705509
// MI455X (gfx1250) — matrix operand order repaired and checked by reading, not compiled, not run on hardware
//
#include <hip/hip_runtime.h>

typedef _Float16 f16;
typedef __attribute__((ext_vector_type(16))) _Float16 v16h;
typedef __attribute__((ext_vector_type(8)))  _Float16 v8h;
typedef __attribute__((ext_vector_type(8)))  float    v8f;
typedef __attribute__((ext_vector_type(4)))  unsigned u32x4;
typedef __attribute__((ext_vector_type(8)))  int      i32x8;
typedef __attribute__((ext_vector_type(4)))  int      i32x4;

#define N_NODES   2048
#define NN        ((size_t)N_NODES * N_NODES)
#define W_IN      512
#define W_OUT     256
#define THRESHOLD 0.05f
#define W_EPS     1e-4f
#define LDS_PITCH 40   /* f16 per LDS tile row: 32 data + 4-dword TDM pad -> 20 dwords */

// ---------------------------------------------------------------------------
// WMMA fragment layouts (ISA 7.12.2 / 7.12.4):
// A (16x32 f16, M=lane&15): lanes<16 hold K 0-7 & 16-23, lanes>=16 K 8-15 & 24-31.
// B (32x16 f16, N=lane&15): lanes<16 hold K 0-7 & 16-23, lanes>=16 K 8-15 & 24-31.
// Matrices stored so each lane's 16 K-values are two contiguous 16B chunks:
// A row-major [M][K], B transposed row-major [N][K].
// ---------------------------------------------------------------------------
__device__ __forceinline__ v16h load_a_frag(const f16* __restrict__ base,
                                            int row, int k0, int half) {
  const f16* p = base + (size_t)row * N_NODES + k0 + 8 * half;
  union { v16h v; v8h c[2]; } u;
  u.c[0] = *(const v8h*)p;
  u.c[1] = *(const v8h*)(p + 16);
  return u.v;
}

__device__ __forceinline__ v16h load_b_frag(const f16* __restrict__ baseT,
                                            int col, int k0, int half) {
  const f16* p = baseT + (size_t)col * N_NODES + k0 + 8 * half;
  union { v16h v; v8h c[2]; } u;
  u.c[0] = *(const v8h*)p;
  u.c[1] = *(const v8h*)(p + 16);
  return u.v;
}

__device__ __forceinline__ v16h lds_a_frag(const f16* rowp, int half) {
  union { v16h v; v8h c[2]; } u;
  u.c[0] = *(const v8h*)(rowp + 8 * half);
  u.c[1] = *(const v8h*)(rowp + 16 + 8 * half);
  return u.v;
}

__device__ __forceinline__ v16h lds_b_frag(const f16* rowp, int half) {
  union { v16h v; v8h c[2]; } u;
  u.c[0] = *(const v8h*)(rowp + 8 * half);
  u.c[1] = *(const v8h*)(rowp + 16 + 8 * half);
  return u.v;
}

#define WMMA_F16(a, b, c) \
  __builtin_amdgcn_wmma_f32_16x16x32_f16(false, (a), false, (b), (short)0, (c), false, false)

// ---------------------------------------------------------------------------
// Tensor Data Mover: 2-D tile load Global->LDS (CDNA5 ISA).
// D# group0: count=1, lds_addr, 57-bit global_addr, type=2.
// D# group1: data_size=2B, pad_enable (4 dwords pad per 16 dwords => 80B rows),
//            tensor_dim0/1 = 2048, tile_dim0 = 32, tile_dim1 = rows,
//            tensor_dim0_stride = 2048.
// Groups 2/3 (and the trailing i32x8 operand of the 6-arg builtin) are zero
// for a 2-D tile.
// ---------------------------------------------------------------------------
__device__ __forceinline__ unsigned lds_addr_of(const void* p) {
  return (unsigned)(unsigned long long)(__attribute__((address_space(3))) const char*)p;
}

__device__ __forceinline__ void tdm_load_tile(unsigned lds_off, const void* gptr,
                                              int tile_rows) {
  unsigned long long ga = (unsigned long long)gptr;
  u32x4 g0;
  g0[0] = 1u;                                   // count=1 (valid user descriptor)
  g0[1] = lds_off;                              // lds_addr (bytes)
  g0[2] = (unsigned)ga;                         // global_addr[31:0]
  g0[3] = (unsigned)((ga >> 32) & 0x01FFFFFFull) | (2u << 30);  // addr[56:32], type=2
  i32x8 g1;
  g1[0] = (int)((1u << 16) |                    // data_size = 2 bytes
                (1u << 20) |                    // pad_enable
                (3u << 22) |                    // pad_interval: 16 dwords
                (3u << 25));                    // pad_amount:   4 dwords
  g1[1] = (int)((unsigned)N_NODES << 16);       // tensor_dim0 = 2048 (low16)
  g1[2] = (int)((unsigned)N_NODES << 16);       // dim0 hi=0 | tensor_dim1 = 2048 (low16)
  g1[3] = (int)(32u << 16);                     // dim1 hi=0 | tile_dim0 = 32
  g1[4] = tile_rows;                            // tile_dim1 | tile_dim2=0
  g1[5] = N_NODES;                              // tensor_dim0_stride low32
  g1[6] = 0;                                    // stride0 hi | stride1 low16
  g1[7] = 0;                                    // stride1 hi
  i32x4 z4 = {0, 0, 0, 0};
  i32x8 z8 = {0, 0, 0, 0, 0, 0, 0, 0};
  __builtin_amdgcn_tensor_load_to_lds(g0, g1, z4, z4, z8, 0);
}

// ---------------------------------------------------------------------------
// Kernel 1: GTConv — softmax over edge types + weighted adjacency sum.
// Writes A1 row-major f16 and A2 transposed f16 (LDS tile transpose).
// ---------------------------------------------------------------------------
__device__ __forceinline__ void softmax4(const float* __restrict__ W, float f[4]) {
  float w0 = W[0], w1 = W[1], w2 = W[2], w3 = W[3];
  float mx = fmaxf(fmaxf(w0, w1), fmaxf(w2, w3));
  float e0 = expf(w0 - mx), e1 = expf(w1 - mx), e2 = expf(w2 - mx), e3 = expf(w3 - mx);
  float inv = 1.0f / (e0 + e1 + e2 + e3);
  f[0] = e0 * inv; f[1] = e1 * inv; f[2] = e2 * inv; f[3] = e3 * inv;
  #pragma unroll
  for (int e = 0; e < 4; ++e) if (f[e] < W_EPS) f[e] = 0.0f;
}

__global__ __launch_bounds__(256) void gtconv_kernel(
    const float* __restrict__ A, const float* __restrict__ W1a,
    const float* __restrict__ W1b, f16* __restrict__ A1h,
    f16* __restrict__ A2Th) {
  float f1[2][4], f2[2][4];
  softmax4(W1a + 0, f1[0]); softmax4(W1a + 4, f1[1]);
  softmax4(W1b + 0, f2[0]); softmax4(W1b + 4, f2[1]);

  __shared__ f16 t2[2][64][66];   // 33-dword rows -> conflict-free transpose reads
  const int i0 = blockIdx.y * 64, j0 = blockIdx.x * 64;

  #pragma unroll 4
  for (int it = 0; it < 16; ++it) {
    int idx = it * 256 + threadIdx.x;
    int r = idx >> 6, cl = idx & 63;
    size_t g = (size_t)(i0 + r) * N_NODES + (j0 + cl);
    float a0 = A[g], a1 = A[NN + g], a2 = A[2 * NN + g], a3 = A[3 * NN + g];
    #pragma unroll
    for (int c = 0; c < 2; ++c) {
      float s1 = f1[c][0] * a0 + f1[c][1] * a1 + f1[c][2] * a2 + f1[c][3] * a3;
      float s2 = f2[c][0] * a0 + f2[c][1] * a1 + f2[c][2] * a2 + f2[c][3] * a3;
      A1h[c * NN + g] = (f16)s1;
      t2[c][r][cl] = (f16)s2;
    }
  }
  __syncthreads();
  #pragma unroll 4
  for (int it = 0; it < 16; ++it) {
    int idx = it * 256 + threadIdx.x;
    int r = idx >> 6, cl = idx & 63;
    size_t g = (size_t)(j0 + r) * N_NODES + (i0 + cl);
    A2Th[g]      = t2[0][cl][r];
    A2Th[NN + g] = t2[1][cl][r];
  }
}

// ---------------------------------------------------------------------------
// Kernel 2: H[c] = A1[c] @ A2[c], threshold > 0.05, store f16.
// Block tile 128x128 (8 waves, wave tile 64x32). K-tiles of 32 staged in LDS
// by the Tensor Data Mover, double-buffered; fragments read via ds_load_b128.
// ---------------------------------------------------------------------------
__global__ __launch_bounds__(256) void gemm1_kernel(
    const f16* __restrict__ A1h, const f16* __restrict__ A2Th,
    f16* __restrict__ Hh) {
  __shared__ f16 As[2][128][LDS_PITCH];
  __shared__ f16 Bs[2][128][LDS_PITCH];

  const int lane = threadIdx.x & 31, wave = threadIdx.x >> 5;
  const int mlo = lane & 15, hh = lane >> 4;
  const int wm = wave >> 2, wn = wave & 3;
  const int ch = blockIdx.z;
  const int m0b = blockIdx.y * 128;
  const int n0b = blockIdx.x * 128;
  const f16* __restrict__ Ab = A1h + (size_t)ch * NN;
  const f16* __restrict__ Bb = A2Th + (size_t)ch * NN;

  if (wave == 0) {
    tdm_load_tile(lds_addr_of(&As[0][0][0]), Ab + (size_t)m0b * N_NODES, 128);
    tdm_load_tile(lds_addr_of(&Bs[0][0][0]), Bb + (size_t)n0b * N_NODES, 128);
    __builtin_amdgcn_s_wait_tensorcnt(0);
  }
  __syncthreads();

  v8f acc[4][2] = {};
  for (int ks = 0; ks < 64; ++ks) {
    const int cur = ks & 1;
    if (wave == 0 && ks + 1 < 64) {
      const int nxt = cur ^ 1, k0 = (ks + 1) * 32;
      tdm_load_tile(lds_addr_of(&As[nxt][0][0]), Ab + (size_t)m0b * N_NODES + k0, 128);
      tdm_load_tile(lds_addr_of(&Bs[nxt][0][0]), Bb + (size_t)n0b * N_NODES + k0, 128);
    }
    v16h a[4], b[2];
    #pragma unroll
    for (int i = 0; i < 4; ++i)
      a[i] = lds_a_frag(&As[cur][wm * 64 + 16 * i + mlo][0], hh);
    #pragma unroll
    for (int j = 0; j < 2; ++j)
      b[j] = lds_b_frag(&Bs[cur][wn * 32 + 16 * j + mlo][0], hh);
    #pragma unroll
    for (int i = 0; i < 4; ++i)
      #pragma unroll
      for (int j = 0; j < 2; ++j)
        acc[i][j] = WMMA_F16(a[i], b[j], acc[i][j]);

    if (wave == 0) __builtin_amdgcn_s_wait_tensorcnt(0);
    __syncthreads();
  }

  #pragma unroll
  for (int i = 0; i < 4; ++i)
    #pragma unroll
    for (int j = 0; j < 2; ++j) {
      int col = n0b + wn * 32 + 16 * j + mlo;
      #pragma unroll
      for (int v = 0; v < 8; ++v) {
        int row = m0b + wm * 64 + 16 * i + v + 8 * hh;
        float val = acc[i][j][v];
        val = val > THRESHOLD ? val : 0.0f;
        Hh[(size_t)ch * NN + (size_t)row * N_NODES + col] = (f16)val;
      }
    }
}

// ---------------------------------------------------------------------------
// Kernel 3: row degrees of thresholded H (deterministic, no atomics).
// ---------------------------------------------------------------------------
__global__ __launch_bounds__(256) void deg_kernel(const f16* __restrict__ Hh,
                                                  float* __restrict__ deg) {
  int wid = (blockIdx.x << 3) + (threadIdx.x >> 5);
  int lane = threadIdx.x & 31;
  int ch = wid >> 11, row = wid & 2047;
  const f16* p = Hh + (size_t)ch * NN + (size_t)row * N_NODES;
  float s = 0.0f;
  #pragma unroll
  for (int base = 0; base < N_NODES; base += 512) {
    v8h v = *(const v8h*)(p + base + lane * 16);
    #pragma unroll
    for (int u = 0; u < 8; ++u) s += (float)v[u];
  }
  #pragma unroll
  for (int off = 16; off > 0; off >>= 1) s += __shfl_xor(s, off, 32);
  if (lane == 0) deg[(ch << 11) + row] = s;
}

// ---------------------------------------------------------------------------
// Kernel 4: XWT[f][j] = sum_k X[j][k] * Wg[k][f]  (transposed for B-operand).
// ---------------------------------------------------------------------------
__global__ __launch_bounds__(256) void xwt_kernel(const float* __restrict__ X,
                                                  const float* __restrict__ Wg,
                                                  f16* __restrict__ XWT) {
  __shared__ float xs[16][W_IN];
  const int j0 = blockIdx.x * 16;
  #pragma unroll 4
  for (int it = 0; it < 32; ++it) {
    int idx = it * 256 + threadIdx.x;
    int r = idx >> 9, c = idx & 511;
    xs[r][c] = X[(size_t)(j0 + r) * W_IN + c];
  }
  __syncthreads();
  const int f = threadIdx.x;
  float acc[16] = {};
  for (int k = 0; k < W_IN; ++k) {
    float wv = Wg[(size_t)k * W_OUT + f];
    #pragma unroll
    for (int jj = 0; jj < 16; ++jj) acc[jj] += xs[jj][k] * wv;
  }
  union { v8h v[2]; f16 e[16]; } u;
  #pragma unroll
  for (int jj = 0; jj < 16; ++jj) u.e[jj] = (f16)acc[jj];
  f16* p = XWT + (size_t)f * N_NODES + j0;
  *(v8h*)p = u.v[0];
  *(v8h*)(p + 8) = u.v[1];
}

// ---------------------------------------------------------------------------
// Kernel 5: out[i][c*256+f] = relu( deg[c][i]^-1 * (H[c] @ XW)[i][f] + bg[f] )
// Direct-global WMMA GEMM (operands are L2-resident).
// ---------------------------------------------------------------------------
__global__ __launch_bounds__(256) void gemm2_kernel(
    const f16* __restrict__ Hh, const f16* __restrict__ XWT,
    const float* __restrict__ deg, const float* __restrict__ bg,
    float* __restrict__ out) {
  const int lane = threadIdx.x & 31, wave = threadIdx.x >> 5;
  const int mlo = lane & 15, hh = lane >> 4;
  const int wm = wave >> 2, wn = wave & 3;
  const int ch = blockIdx.z;
  const int m0 = blockIdx.y * 128 + wm * 64;
  const int n0 = blockIdx.x * 128 + wn * 32;
  const f16* __restrict__ Ab = Hh + (size_t)ch * NN;

  v8f acc[4][2] = {};
  for (int k0 = 0; k0 < N_NODES; k0 += 32) {
    v16h a[4], b[2];
    #pragma unroll
    for (int i = 0; i < 4; ++i) a[i] = load_a_frag(Ab, m0 + 16 * i + mlo, k0, hh);
    #pragma unroll
    for (int j = 0; j < 2; ++j) b[j] = load_b_frag(XWT, n0 + 16 * j + mlo, k0, hh);
    #pragma unroll
    for (int i = 0; i < 4; ++i)
      #pragma unroll
      for (int j = 0; j < 2; ++j)
        acc[i][j] = WMMA_F16(a[i], b[j], acc[i][j]);
  }

  float bgv[2];
  #pragma unroll
  for (int j = 0; j < 2; ++j) bgv[j] = bg[n0 + 16 * j + mlo];

  #pragma unroll
  for (int i = 0; i < 4; ++i)
    #pragma unroll
    for (int v = 0; v < 8; ++v) {
      int row = m0 + 16 * i + v + 8 * hh;
      float dv = deg[(ch << 11) + row];
      float dinv = dv > 0.0f ? 1.0f / dv : 0.0f;
      #pragma unroll
      for (int j = 0; j < 2; ++j) {
        int col = n0 + 16 * j + mlo;
        float val = acc[i][j][v] * dinv + bgv[j];
        val = val > 0.0f ? val : 0.0f;
        out[(size_t)row * (2 * W_OUT) + (ch << 8) + col] = val;
      }
    }
}

// ---------------------------------------------------------------------------
extern "C" void kernel_launch(void* const* d_in, const int* in_sizes, int n_in,
                              void* d_out, int out_size, void* d_ws, size_t ws_size,
                              hipStream_t stream) {
  const float* A   = (const float*)d_in[0];   // [4, 2048, 2048]
  const float* X   = (const float*)d_in[1];   // [2048, 512]
  const float* W1a = (const float*)d_in[2];   // [2, 4]
  const float* W1b = (const float*)d_in[3];   // [2, 4]
  const float* Wg  = (const float*)d_in[4];   // [512, 256]
  const float* bg  = (const float*)d_in[5];   // [256]
  float* out = (float*)d_out;                 // [2048, 512]

  char* w = (char*)d_ws;
  const size_t szMatH = 2 * NN * sizeof(f16);             // 16 MB each
  f16*   A1h  = (f16*)(w);
  f16*   A2Th = (f16*)(w + szMatH);
  f16*   Hh   = (f16*)(w + 2 * szMatH);
  f16*   XWT  = (f16*)(w + 3 * szMatH);                   // 1 MB
  float* deg  = (float*)(w + 3 * szMatH + (size_t)W_OUT * N_NODES * sizeof(f16));

  gtconv_kernel<<<dim3(32, 32, 1), 256, 0, stream>>>(A, W1a, W1b, A1h, A2Th);
  gemm1_kernel<<<dim3(16, 16, 2), 256, 0, stream>>>(A1h, A2Th, Hh);
  deg_kernel<<<dim3(512, 1, 1), 256, 0, stream>>>(Hh, deg);
  xwt_kernel<<<dim3(128, 1, 1), 256, 0, stream>>>(X, Wg, XWT);
  gemm2_kernel<<<dim3(2, 16, 2), 256, 0, stream>>>(Hh, XWT, deg, bg, out);
}
